// FixedReservoir_73254962201248
// MI455X (gfx1250) — matrix operand order repaired and checked by reading, not compiled, not run on hardware
//
#include <hip/hip_runtime.h>
#include <math.h>

// Problem constants (from reference)
constexpr int B_   = 256;
constexpr int T_   = 64;
constexpr int N_   = 4096;
constexpr int DIN_ = 128;
#define ALPHA_F 0.17f

// GEMM blocking: block = 256 threads = 8 waves arranged 4(M) x 2(N).
// Block output tile: 64 (M) x 128 (N). Wave output: 16 (M) x 64 (N).
constexpr int BM = 64;
constexpr int BN = 128;
constexpr int BK = 32;
constexpr int LDS_STRIDE = 40;  // 32 k-elements + 8 pad (80B rows -> conflict-free)

typedef __attribute__((ext_vector_type(16))) __bf16 v16bf;
typedef __attribute__((ext_vector_type(8)))  __bf16 v8bf;
typedef __attribute__((ext_vector_type(8)))  float  v8f;
typedef __attribute__((ext_vector_type(4)))  int    v4i;

typedef __attribute__((address_space(1))) v4i gv4i;  // global int4
typedef __attribute__((address_space(3))) v4i sv4i;  // LDS int4

// ---- CDNA5 async global->LDS staging (no VGPR round-trip), with fallback ----
#if defined(__gfx1250__) && __has_builtin(__builtin_amdgcn_global_load_async_to_lds_b128)
#define HAVE_ASYNC_LDS 1
#else
#define HAVE_ASYNC_LDS 0
#endif

__device__ __forceinline__ void async_cp16(const __bf16* g, __bf16* l) {
#if HAVE_ASYNC_LDS
  __builtin_amdgcn_global_load_async_to_lds_b128((gv4i*)g, (sv4i*)l, 0, 0);
#else
  *(v8bf*)l = *(const v8bf*)g;   // sync copy via regs (known-good path)
#endif
}

template <int N>
__device__ __forceinline__ void wait_async() {
#if HAVE_ASYNC_LDS
#if __has_builtin(__builtin_amdgcn_s_wait_asynccnt)
  __builtin_amdgcn_s_wait_asynccnt(N);
#else
  asm volatile("s_wait_asynccnt %0" :: "i"(N) : "memory");
#endif
#endif
}

// Build a 16-element bf16 vector from two 8-element (16B) loads.
__device__ __forceinline__ v16bf ld_pair(const __bf16* lo, const __bf16* hi) {
  v8bf l = *(const v8bf*)lo;
  v8bf h = *(const v8bf*)hi;
  return __builtin_shufflevector(l, h, 0, 1, 2, 3, 4, 5, 6, 7,
                                       8, 9, 10, 11, 12, 13, 14, 15);
}

__global__ void f32_to_bf16_kernel(const float* __restrict__ src,
                                   __bf16* __restrict__ dst, int n) {
  int i = blockIdx.x * blockDim.x + threadIdx.x;
  if (i < n) dst[i] = (__bf16)src[i];
}

__global__ void zero_state_kernel(float* __restrict__ vf,
                                  __bf16* __restrict__ vb, int n) {
  int i = blockIdx.x * blockDim.x + threadIdx.x;
  if (i < n) { vf[i] = 0.0f; vb[i] = (__bf16)0.0f; }
}

// One reservoir timestep:
//   pre   = V @ W^T + u[:,t,:] @ W_in^T        (bf16 WMMA, fp32 accumulate)
//   V_new = (1-a)*V + a*tanh(pre)
__launch_bounds__(256)
__global__ void reservoir_step_kernel(
    const __bf16* __restrict__ Vb_in,   // [B,N] bf16 state
    const float*  __restrict__ Vf_in,   // [B,N] fp32 state
    const __bf16* __restrict__ Wb,      // [N,N] bf16, row-major (n,k)
    const __bf16* __restrict__ Winb,    // [N,DIN] bf16, row-major (n,d)
    const __bf16* __restrict__ ub,      // [B,T,DIN] bf16
    int t,
    float*  __restrict__ Vf_out,        // [B,N]
    __bf16* __restrict__ Vb_out,        // [B,N]
    float*  __restrict__ out)           // [B,T,N]
{
  // Double-buffered staging tiles: A = V rows (BM x BK), B = W rows (BN x BK).
  __shared__ __bf16 sA[2][BM * LDS_STRIDE];   // 2 * 64*40*2B  = 10.0 KB
  __shared__ __bf16 sB[2][BN * LDS_STRIDE];   // 2 * 128*40*2B = 20.0 KB

  const int tid  = threadIdx.x;
  const int lane = tid & 31;
  const int wave = tid >> 5;
  const int half = lane >> 4;   // 0: lanes 0-15, 1: lanes 16-31
  const int lcol = lane & 15;

  const int wm = wave & 3;      // M-tile within block (0..3)
  const int wn = wave >> 2;     // N 64-chunk within block (0..1)
  const int mBase = blockIdx.y * BM;
  const int nBase = blockIdx.x * BN;
  const int mbase = mBase + wm * 16;
  const int nbase = nBase + wn * 64;

  // ---- staging assignment: 256 threads move 768 8-elem chunks per K-step ----
  // thread -> 1 A chunk + 2 B chunks, all at k8 = (tid&3)*8 within the K-slab.
  const int srow = tid >> 2;          // 0..63
  const int sk8  = (tid & 3) * 8;     // 0,8,16,24
  const __bf16* pga  = Vb_in + (size_t)(mBase + srow)      * N_ + sk8;
  const __bf16* pgb0 = Wb    + (size_t)(nBase + srow)      * N_ + sk8;
  const __bf16* pgb1 = Wb    + (size_t)(nBase + 64 + srow) * N_ + sk8;
  const int dA  = srow * LDS_STRIDE + sk8;
  const int dB1 = (64 + srow) * LDS_STRIDE + sk8;

  __bf16* wAd[2]  = {&sA[0][dA],  &sA[1][dA]};
  __bf16* wB0d[2] = {&sB[0][dA],  &sB[1][dA]};   // dB0 == dA layout-wise
  __bf16* wB1d[2] = {&sB[0][dB1], &sB[1][dB1]};

  // ---- per-wave LDS read pointers (WMMA fragment layout) ----
  // A: lane -> row (mrow), elems [half*8 .. +7] and [16+half*8 .. +7]
  const __bf16* la = &sA[0][(wm * 16 + lcol) * LDS_STRIDE + half * 8];
  // B: lane -> col, elems [half*8 .. +7] and [16+half*8 .. +7]
  const int lbOff[4] = {
      (wn * 64 +  0 + lcol) * LDS_STRIDE + half * 8,
      (wn * 64 + 16 + lcol) * LDS_STRIDE + half * 8,
      (wn * 64 + 32 + lcol) * LDS_STRIDE + half * 8,
      (wn * 64 + 48 + lcol) * LDS_STRIDE + half * 8};

  v8f acc[4] = {v8f{}, v8f{}, v8f{}, v8f{}};

  constexpr int KSTEPS = N_ / BK;   // 128

  // Prologue: stage K-slab 0 into buffer 0.
  async_cp16(pga,  wAd[0]);
  async_cp16(pgb0, wB0d[0]);
  async_cp16(pgb1, wB1d[0]);

#pragma unroll 2
  for (int kk = 0; kk < KSTEPS; ++kk) {
    const int buf = kk & 1;

    // Refill the other buffer (all waves finished reading it at the trailing
    // barrier of iteration kk-1), then retire this buffer's 3 copies.
    if (kk + 1 < KSTEPS) {
      const int k2 = (kk + 1) * BK;
      async_cp16(pga  + k2, wAd[buf ^ 1]);
      async_cp16(pgb0 + k2, wB0d[buf ^ 1]);
      async_cp16(pgb1 + k2, wB1d[buf ^ 1]);
      wait_async<3>();
    } else {
      wait_async<0>();
    }
    __syncthreads();

    const int bo = buf * (BM * LDS_STRIDE);  // sA buffer offset in elements
    v16bf a = ld_pair(la + bo, la + bo + 16);
#pragma unroll
    for (int j = 0; j < 4; ++j) {
      const __bf16* lb = &sB[buf][lbOff[j]];
      v16bf b = ld_pair(lb, lb + 16);
      acc[j] = __builtin_amdgcn_wmma_f32_16x16x32_bf16(
          false, a, false, b, (short)0, acc[j], false, false);
    }
    __syncthreads();
  }

  // Folded input projection: K = 128 over W_in with A from u[:,t,:] (direct loads)
  const __bf16* pau = ub + ((size_t)(mbase + lcol) * T_ + t) * DIN_ + half * 8;
  const __bf16* pu[4];
#pragma unroll
  for (int j = 0; j < 4; ++j)
    pu[j] = Winb + (size_t)(nbase + j * 16 + lcol) * DIN_ + half * 8;

#pragma unroll
  for (int k = 0; k < DIN_; k += 32) {
    v16bf a = ld_pair(pau + k, pau + k + 16);
#pragma unroll
    for (int j = 0; j < 4; ++j) {
      v16bf b = ld_pair(pu[j] + k, pu[j] + k + 16);
      acc[j] = __builtin_amdgcn_wmma_f32_16x16x32_bf16(
          false, a, false, b, (short)0, acc[j], false, false);
    }
  }

  // Epilogue: C/D layout -> VGPR r holds (M = r + half*8, N = lcol).
#pragma unroll
  for (int j = 0; j < 4; ++j) {
    const int col = nbase + j * 16 + lcol;
#pragma unroll
    for (int r = 0; r < 8; ++r) {
      const int row = mbase + half * 8 + r;
      const float pre  = acc[j][r];
      const size_t vi  = (size_t)row * N_ + col;
      const float vnew = (1.0f - ALPHA_F) * Vf_in[vi] + ALPHA_F * tanhf(pre);
      Vf_out[vi] = vnew;
      Vb_out[vi] = (__bf16)vnew;
      out[((size_t)row * T_ + t) * (size_t)N_ + col] = vnew;
    }
  }
}

extern "C" void kernel_launch(void* const* d_in, const int* in_sizes, int n_in,
                              void* d_out, int out_size, void* d_ws, size_t ws_size,
                              hipStream_t stream) {
  const float* u_seq = (const float*)d_in[0];  // [B,T,DIN]
  const float* W_in  = (const float*)d_in[1];  // [N,DIN]
  const float* W     = (const float*)d_in[2];  // [N,N]
  float* out = (float*)d_out;                  // [B,T,N]

  // Carve workspace (~52 MB): bf16 copies + ping-pong state.
  char* ws = (char*)d_ws;
  size_t off = 0;
  auto carve = [&](size_t bytes) -> void* {
    void* p = ws + off;
    off += (bytes + 255) & ~(size_t)255;
    return p;
  };
  __bf16* Wb   = (__bf16*)carve((size_t)N_ * N_ * 2);
  __bf16* Winb = (__bf16*)carve((size_t)N_ * DIN_ * 2);
  __bf16* ub   = (__bf16*)carve((size_t)B_ * T_ * DIN_ * 2);
  float*  Vf[2];
  __bf16* Vb[2];
  Vf[0] = (float*)carve((size_t)B_ * N_ * 4);
  Vf[1] = (float*)carve((size_t)B_ * N_ * 4);
  Vb[0] = (__bf16*)carve((size_t)B_ * N_ * 2);
  Vb[1] = (__bf16*)carve((size_t)B_ * N_ * 2);

  const int thr = 256;
  f32_to_bf16_kernel<<<(N_ * N_ + thr - 1) / thr, thr, 0, stream>>>(W, Wb, N_ * N_);
  f32_to_bf16_kernel<<<(N_ * DIN_ + thr - 1) / thr, thr, 0, stream>>>(W_in, Winb, N_ * DIN_);
  f32_to_bf16_kernel<<<(B_ * T_ * DIN_ + thr - 1) / thr, thr, 0, stream>>>(u_seq, ub, B_ * T_ * DIN_);
  zero_state_kernel<<<(B_ * N_ + thr - 1) / thr, thr, 0, stream>>>(Vf[0], Vb[0], B_ * N_);

  // 32 N-blocks x 4 M-blocks = 128 blocks per step.
  dim3 grid(N_ / BN, B_ / BM);
  for (int t = 0; t < T_; ++t) {
    const int s = t & 1, d = s ^ 1;
    reservoir_step_kernel<<<grid, 256, 0, stream>>>(
        Vb[s], Vf[s], Wb, Winb, ub, t, Vf[d], Vb[d], out);
  }
}
